// CouplingHead_46316927320415
// MI455X (gfx1250) — matrix operand order repaired and checked by reading, not compiled, not run on hardware
//
#include <hip/hip_runtime.h>
#include <hip/hip_bf16.h>

typedef __attribute__((ext_vector_type(16))) _Float16 v16h;
typedef __attribute__((ext_vector_type(8)))  float    v8f;

#define BATCH   8
#define NPTS    2048
#define DFEAT   256
#define DHEAD   64
#define MBLK    32
#define LSTRIDE 34   // LDS row stride (floats); 8*34 % 64 == 16 -> the two C half-tiles hit disjoint banks

// ---------------------------------------------------------------------------
// Kernel A: fc = [f | log_eps] ; q = fc@Wq / 8 ; k = fc@Wk  (fp32 math, f16 out)
// One block handles 64 consecutive rows of one batch. Wq/Wk live in LDS.
// ---------------------------------------------------------------------------
__global__ void __launch_bounds__(256)
proj_kernel(const float* __restrict__ f, const float* __restrict__ log_eps,
            const float* __restrict__ Wq, const float* __restrict__ Wk,
            _Float16* __restrict__ qh, _Float16* __restrict__ kh)
{
    extern __shared__ float lds[];
    float* wq = lds;                      // 257*64
    float* wk = wq + 257 * DHEAD;         // 257*64
    float* fs = wk + 257 * DHEAD;         // 64*256

    const int tid = threadIdx.x;
    for (int i = tid; i < 257 * DHEAD; i += 256) { wq[i] = Wq[i]; wk[i] = Wk[i]; }

    const int r0 = blockIdx.x * 64;       // first global row (b*N + n)
    const int b  = r0 >> 11;              // /2048 (64 divides 2048 -> same b in block)
    for (int i = tid; i < 64 * DFEAT; i += 256)
        fs[i] = f[(size_t)r0 * DFEAT + i];
    __syncthreads();

    const float le = log_eps[b];
    const int h  = tid & 63;
    const int rl = tid >> 6;              // 4 row-groups

    for (int rr = rl; rr < 64; rr += 4) {
        const float* fr = fs + rr * DFEAT;
        float aq = le * wq[DFEAT * DHEAD + h];
        float ak = le * wk[DFEAT * DHEAD + h];
        for (int ff = 0; ff < DFEAT; ++ff) {
            const float v = fr[ff];
            aq += v * wq[ff * DHEAD + h];
            ak += v * wk[ff * DHEAD + h];
        }
        const size_t o = (size_t)(r0 + rr) * DHEAD + h;
        qh[o] = (_Float16)(aq * 0.125f);  // fold 1/sqrt(64) into q
        kh[o] = (_Float16)ak;
    }
}

// ---------------------------------------------------------------------------
// Kernel B: per (b, 32-column m-block): logits tile (2048 x 32) via WMMA into
// LDS, column softmax over n, single-pass write of pi and y accumulation.
// 256 threads = 8 waves; wave w: m_sub = (w&1)*16, n-range = (w>>1)*512.
// ---------------------------------------------------------------------------
__global__ void __launch_bounds__(256)
attn_kernel(const _Float16* __restrict__ qh, const _Float16* __restrict__ kh,
            const float* __restrict__ x_tilde,
            float* __restrict__ y_out, float* __restrict__ pi_out)
{
    extern __shared__ float lds[];
    float* Ltile  = lds;                         // NPTS * LSTRIDE
    float* colmax = lds + NPTS * LSTRIDE;        // 32
    float* colsum = colmax + MBLK;               // 32
    float* red    = colsum + MBLK;               // 256
    float* ypart  = red + 256;                   // 32*8*3

    const int b  = blockIdx.x >> 6;              // 64 m-blocks per batch
    const int m0 = (blockIdx.x & 63) * MBLK;

    const int tid  = threadIdx.x;
    const int lane = tid & 31;
    const int wv   = tid >> 5;
    const int g    = lane >> 4;                  // lane group (0/1)
    const int ln   = lane & 15;

    const _Float16* qb = qh + (size_t)b * NPTS * DHEAD;
    const _Float16* kb = kh + (size_t)b * NPTS * DHEAD;

    const int m_sub = (wv & 1) << 4;
    const int nbase = (wv >> 1) << 9;            // * 512

    // B fragments (k^T), invariant over the n loop.
    // Contiguous k order (same as A): B element e (0..15) of lane -> K = 16*g + e, N = ln.
    v16h bf0, bf1;
    {
        const _Float16* kr = kb + (size_t)(m0 + m_sub + ln) * DHEAD + (g << 4);
        ((uint4*)&bf0)[0] = *(const uint4*)(kr);
        ((uint4*)&bf0)[1] = *(const uint4*)(kr + 8);
        ((uint4*)&bf1)[0] = *(const uint4*)(kr + 32);
        ((uint4*)&bf1)[1] = *(const uint4*)(kr + 40);
    }

    // GEMM: 32 n-tiles of 16 rows, K=64 via two chained 16x16x32 WMMAs.
    for (int t = 0; t < 32; ++t) {
        const int n0 = nbase + (t << 4);
        // A layout: element e -> K = 16*g + e, row M = ln.
        const _Float16* qr = qb + (size_t)(n0 + ln) * DHEAD + (g << 4);
        v16h a0, a1;
        ((uint4*)&a0)[0] = *(const uint4*)(qr);
        ((uint4*)&a0)[1] = *(const uint4*)(qr + 8);
        ((uint4*)&a1)[0] = *(const uint4*)(qr + 32);
        ((uint4*)&a1)[1] = *(const uint4*)(qr + 40);

        v8f c = {};
        c = __builtin_amdgcn_wmma_f32_16x16x32_f16(false, a0, false, bf0,
                                                   (short)0, c, false, false);
        c = __builtin_amdgcn_wmma_f32_16x16x32_f16(false, a1, false, bf1,
                                                   (short)0, c, false, false);
        // C layout: VGPR i -> M = i + 8*g, N = ln.
        #pragma unroll
        for (int i = 0; i < 8; ++i)
            Ltile[(n0 + (g << 3) + i) * LSTRIDE + m_sub + ln] = c[i];
    }
    __syncthreads();

    // Column softmax over n (axis=1). Thread (mi, ng): column mi, rows n=ng step 8.
    const int mi = tid & 31;                     // wave-contiguous -> coalesced pi writes
    const int ng = tid >> 5;

    float mx = -3.4e38f;
    for (int n = ng; n < NPTS; n += 8)
        mx = fmaxf(mx, Ltile[n * LSTRIDE + mi]);
    red[mi * 8 + ng] = mx;
    __syncthreads();
    if (ng == 0) {
        float m = red[mi * 8];
        #pragma unroll
        for (int j = 1; j < 8; ++j) m = fmaxf(m, red[mi * 8 + j]);
        colmax[mi] = m;
    }
    __syncthreads();
    const float cm = colmax[mi];

    float s = 0.f;
    for (int n = ng; n < NPTS; n += 8)
        s += __expf(Ltile[n * LSTRIDE + mi] - cm);
    red[mi * 8 + ng] = s;
    __syncthreads();
    if (ng == 0) {
        float t = 0.f;
        #pragma unroll
        for (int j = 0; j < 8; ++j) t += red[mi * 8 + j];
        colsum[mi] = t;
    }
    __syncthreads();
    const float inv = 1.0f / colsum[mi];

    // Single streaming pass: write pi (coalesced 128B rows) + accumulate y.
    float y0 = 0.f, y1 = 0.f, y2 = 0.f;
    float* piB = pi_out + (size_t)b * NPTS * NPTS + m0 + mi;
    const float* xb = x_tilde + (size_t)b * NPTS * 3;
    for (int n = ng; n < NPTS; n += 8) {
        const float p = __expf(Ltile[n * LSTRIDE + mi] - cm) * inv;
        piB[(size_t)n * NPTS] = p * (1.0f / (float)NPTS);
        const float x0 = xb[n * 3 + 0];
        const float x1 = xb[n * 3 + 1];
        const float x2 = xb[n * 3 + 2];
        y0 += p * x0; y1 += p * x1; y2 += p * x2;
    }
    float* yp = ypart + (mi * 8 + ng) * 3;
    yp[0] = y0; yp[1] = y1; yp[2] = y2;
    __syncthreads();
    if (ng == 0) {
        float s0 = 0.f, s1 = 0.f, s2 = 0.f;
        #pragma unroll
        for (int j = 0; j < 8; ++j) {
            s0 += ypart[(mi * 8 + j) * 3 + 0];
            s1 += ypart[(mi * 8 + j) * 3 + 1];
            s2 += ypart[(mi * 8 + j) * 3 + 2];
        }
        float* yo = y_out + ((size_t)b * NPTS + m0 + mi) * 3;
        yo[0] = s0; yo[1] = s1; yo[2] = s2;
    }
}

// ---------------------------------------------------------------------------
extern "C" void kernel_launch(void* const* d_in, const int* in_sizes, int n_in,
                              void* d_out, int out_size, void* d_ws, size_t ws_size,
                              hipStream_t stream)
{
    const float* f       = (const float*)d_in[0];   // (8,2048,256)
    const float* x_tilde = (const float*)d_in[1];   // (8,2048,3)
    const float* log_eps = (const float*)d_in[2];   // (8,)
    const float* Wq      = (const float*)d_in[3];   // (257,64)
    const float* Wk      = (const float*)d_in[4];   // (257,64)

    float* y_out  = (float*)d_out;                              // (8,2048,3)
    float* pi_out = y_out + (size_t)BATCH * NPTS * 3;           // (8,2048,2048)

    _Float16* qh = (_Float16*)d_ws;                             // 2 MB
    _Float16* kh = qh + (size_t)BATCH * NPTS * DHEAD;           // 2 MB

    const size_t ldsA = (size_t)(257 * DHEAD * 2 + 64 * DFEAT) * sizeof(float);
    const size_t ldsB = (size_t)(NPTS * LSTRIDE + MBLK + MBLK + 256 + MBLK * 8 * 3) * sizeof(float);

    hipFuncSetAttribute((const void*)proj_kernel,
                        hipFuncAttributeMaxDynamicSharedMemorySize, (int)ldsA);
    hipFuncSetAttribute((const void*)attn_kernel,
                        hipFuncAttributeMaxDynamicSharedMemorySize, (int)ldsB);

    proj_kernel<<<(BATCH * NPTS) / 64, 256, ldsA, stream>>>(f, log_eps, Wq, Wk, qh, kh);
    attn_kernel<<<BATCH * (NPTS / MBLK), 256, ldsB, stream>>>(qh, kh, x_tilde, y_out, pi_out);
}
